// Net_19335942767008
// MI455X (gfx1250) — matrix operand order repaired and checked by reading, not compiled, not run on hardware
//
#include <hip/hip_runtime.h>
#include <hip/hip_bf16.h>
#include <cstdint>

typedef __bf16 bf16_t;
typedef __attribute__((ext_vector_type(16))) __bf16 v16bf;
typedef __attribute__((ext_vector_type(8)))  float  v8f;

static __device__ __forceinline__ v8f v8f_zero() {
  v8f z;
#pragma unroll
  for (int i = 0; i < 8; ++i) z[i] = 0.0f;
  return z;
}

static __device__ __forceinline__ v8f wmma_bf16(v16bf a, v16bf b, v8f c) {
  // D = A(16x32 bf16) x B(32x16 bf16) + C(16x16 f32)
  return __builtin_amdgcn_wmma_f32_16x16x32_bf16(false, a, false, b, (short)0, c,
                                                 false, false);
}

// K index of element e for the 16-bit A fragment (16x32, wave32 layout)
static __device__ __forceinline__ int a_kmap(int lane, int e) {
  return e + ((e >= 8) ? 8 : 0) + ((lane >= 16) ? 8 : 0);
}
// K index of element e for the 16-bit B fragment (32x16, wave32 layout)
static __device__ __forceinline__ int b_kmap(int lane, int e) {
  return e + ((e >= 8) ? 8 : 0) + ((lane >= 16) ? 8 : 0);
}

// pack a float4 into 4 consecutive bf16 fragment elements
static __device__ __forceinline__ void put4(v16bf& d, int base, float4 f) {
  d[base + 0] = (bf16_t)f.x;
  d[base + 1] = (bf16_t)f.y;
  d[base + 2] = (bf16_t)f.z;
  d[base + 3] = (bf16_t)f.w;
}

// A fragment from a row-major f32 row (K multiple of 32): two contiguous
// 8-float runs per lane -> b128 loads.
static __device__ __forceinline__ v16bf load_a_frag(const float* rowk0, int lane) {
  const float4* ap = (const float4*)(rowk0 + ((lane >= 16) ? 8 : 0));
  v16bf a;
  put4(a, 0, ap[0]);   // k = base+0..3
  put4(a, 4, ap[1]);   // k = base+4..7
  put4(a, 8, ap[4]);   // k = base+16..19
  put4(a, 12, ap[5]);  // k = base+20..23
  return a;
}

// B fragment from a row-major f32 weight row (K multiple of 32): two
// contiguous 8-float runs per lane -> b128 loads.
static __device__ __forceinline__ v16bf load_b_frag(const float* rowk0, int lane) {
  const float4* bp = (const float4*)(rowk0 + ((lane >= 16) ? 8 : 0));
  v16bf b;
  put4(b, 0, bp[0]);
  put4(b, 4, bp[1]);
  put4(b, 8, bp[4]);
  put4(b, 12, bp[5]);
  return b;
}

// -------------------------------------------------------------------------
// Fused conv3x3(SAME) + bias + eval-BN + ReLU + maxpool2x2 as im2col WMMA.
// One wave handles IPW images; M-rows = IPW*HIN*HIN (must be multiple of 16).
// -------------------------------------------------------------------------
template <int CIN, int COUT, int HIN, int IPW>
__global__ void __launch_bounds__(128)
conv_bn_relu_pool(const float* __restrict__ in,    // [nimg, CIN, HIN, HIN]
                  const float* __restrict__ w,     // [COUT, CIN, 3, 3]
                  const float* __restrict__ bias,  // [COUT]
                  const float* __restrict__ gamma, // [COUT]
                  const float* __restrict__ beta,  // [COUT]
                  float* __restrict__ out,         // [nimg, COUT, HIN/2, HIN/2]
                  int nimg) {
  constexpr int WAVES  = 4;
  constexpr int HW     = HIN * HIN;
  constexpr int HO     = HIN / 2;
  constexpr int K      = CIN * 9;
  constexpr int KSTEPS = (K + 31) / 32;
  constexpr int MROWS  = IPW * HW;    // rows per wave
  constexpr int MTILES = MROWS / 16;
  constexpr int NTILES = COUT / 16;
  constexpr int IPB    = WAVES * IPW; // images per block

  __shared__ float s_in[IPB][CIN * HW];
  __shared__ float s_out[WAVES][MROWS * COUT];

  const int tid  = threadIdx.x;
  const int wave = tid >> 5;
  const int lane = tid & 31;
  const int img0 = blockIdx.x * IPB;

  // cooperative, coalesced image staging
  for (int i = tid; i < IPB * CIN * HW; i += blockDim.x) {
    int im = i / (CIN * HW);
    int r  = i % (CIN * HW);
    int g  = img0 + im;
    s_in[im][r] = (g < nimg) ? in[(size_t)g * (CIN * HW) + r] : 0.0f;
  }
  __syncthreads();

  for (int mt = 0; mt < MTILES; ++mt) {
    v8f acc[NTILES];
#pragma unroll
    for (int nt = 0; nt < NTILES; ++nt) acc[nt] = v8f_zero();

    const int mrow = mt * 16 + (lane & 15);
    const int imL  = wave * IPW + mrow / HW; // block-local image
    const int pix  = mrow % HW;
    const int py   = pix / HIN;
    const int px   = pix % HIN;

    for (int ks = 0; ks < KSTEPS; ++ks) {
      v16bf afrag; // im2col gather from LDS, zero-padded borders & K tail
#pragma unroll
      for (int e = 0; e < 16; ++e) {
        int k = ks * 32 + a_kmap(lane, e);
        float v = 0.0f;
        if (k < K) {
          int c = k / 9, t = k % 9;
          int yy = py + t / 3 - 1;
          int xx = px + t % 3 - 1;
          if (yy >= 0 && yy < HIN && xx >= 0 && xx < HIN)
            v = s_in[imL][(c * HIN + yy) * HIN + xx];
        }
        afrag[e] = (bf16_t)v;
      }
#pragma unroll
      for (int nt = 0; nt < NTILES; ++nt) {
        const int n = nt * 16 + (lane & 15);
        v16bf bfrag; // OIHW row n is contiguous in exactly im2col-k order
        if constexpr ((K % 32) == 0) {
          bfrag = load_b_frag(w + (size_t)n * K + ks * 32, lane); // b128 path
        } else {
#pragma unroll
          for (int e = 0; e < 16; ++e) {
            int k = ks * 32 + b_kmap(lane, e);
            bfrag[e] = (bf16_t)((k < K) ? w[n * K + k] : 0.0f);
          }
        }
        acc[nt] = wmma_bf16(afrag, bfrag, acc[nt]);
      }
    }
    // epilogue: bias + BN(scale=g/sqrt(1+eps), shift=beta) + ReLU -> LDS
#pragma unroll
    for (int nt = 0; nt < NTILES; ++nt) {
      const int n  = nt * 16 + (lane & 15);
      const float sc = gamma[n] * 0.99999500003749968f; // 1/sqrt(1+1e-5)
      const float bb = beta[n];
      const float bi = bias[n];
#pragma unroll
      for (int v = 0; v < 8; ++v) {
        int m = mt * 16 + v + ((lane >= 16) ? 8 : 0);
        float val = (acc[nt][v] + bi) * sc + bb;
        s_out[wave][m * COUT + n] = fmaxf(val, 0.0f);
      }
    }
  }

  // maxpool 2x2 stride 2 from LDS (wave-private region; LDS ops in-order)
  for (int i = lane; i < IPW * COUT * HO * HO; i += 32) {
    int limg = i / (COUT * HO * HO);
    int r    = i % (COUT * HO * HO);
    int c    = r / (HO * HO);
    int op   = r % (HO * HO);
    int oy = op / HO, ox = op % HO;
    int mbase = limg * HW + (2 * oy) * HIN + 2 * ox;
    float v00 = s_out[wave][(mbase) * COUT + c];
    float v01 = s_out[wave][(mbase + 1) * COUT + c];
    float v10 = s_out[wave][(mbase + HIN) * COUT + c];
    float v11 = s_out[wave][(mbase + HIN + 1) * COUT + c];
    float mx  = fmaxf(fmaxf(v00, v01), fmaxf(v10, v11));
    int g = img0 + wave * IPW + limg;
    if (g < nimg) out[((size_t)g * COUT + c) * (HO * HO) + op] = mx;
  }
}

// -------------------------------------------------------------------------
// C[M,N] = act(A[M,K] @ W[N,K]^T + bias); one wave = 16 rows x full N.
// Fragment operands are contiguous per lane -> b128 global loads.
// -------------------------------------------------------------------------
template <int K, int N, bool RELU>
__global__ void __launch_bounds__(128)
gemm_rowmajor_wt(const float* __restrict__ A, const float* __restrict__ W,
                 const float* __restrict__ bias, float* __restrict__ C, int M) {
  constexpr int NT = N / 16;
  constexpr int KS = K / 32;
  const int lane = threadIdx.x & 31;
  const int wave = threadIdx.x >> 5;
  const int m0   = (blockIdx.x * (blockDim.x >> 5) + wave) * 16;
  if (m0 >= M) return; // wave-uniform

  v8f acc[NT];
#pragma unroll
  for (int nt = 0; nt < NT; ++nt) acc[nt] = v8f_zero();

  const int mr = m0 + (lane & 15);
  for (int ks = 0; ks < KS; ++ks) {
    const float* arow = A + (size_t)mr * K + ks * 32;
    if (ks + 1 < KS) __builtin_prefetch(arow + 32, 0, 0); // global_prefetch_b8
    v16bf afrag = load_a_frag(arow, lane);
#pragma unroll
    for (int nt = 0; nt < NT; ++nt) {
      const int n = nt * 16 + (lane & 15);
      v16bf bfrag = load_b_frag(W + (size_t)n * K + ks * 32, lane);
      acc[nt] = wmma_bf16(afrag, bfrag, acc[nt]);
    }
  }
#pragma unroll
  for (int nt = 0; nt < NT; ++nt) {
    const int n   = nt * 16 + (lane & 15);
    const float b = bias ? bias[n] : 0.0f;
#pragma unroll
    for (int v = 0; v < 8; ++v) {
      int m     = m0 + v + ((lane >= 16) ? 8 : 0);
      float val = acc[nt][v] + b;
      if (RELU) val = fmaxf(val, 0.0f);
      C[(size_t)m * N + n] = val;
    }
  }
}

// -------------------------------------------------------------------------
// dinv[i] = rsqrt(1 + sum_j gso[b,i,j])  (self-loop degree)
// -------------------------------------------------------------------------
__global__ void dinv_kernel(const float* __restrict__ gso, float* __restrict__ dinv) {
  int i = blockIdx.x * blockDim.x + threadIdx.x; // exact grid = B*64
  const float4* row = (const float4*)(gso + (size_t)i * 64);
  float s = 1.0f;
#pragma unroll
  for (int j = 0; j < 16; ++j) {
    float4 v = row[j];
    s += (v.x + v.y) + (v.z + v.w);
  }
  dinv[i] = __frsqrt_rn(s);
}

// -------------------------------------------------------------------------
// H[b] = relu(Ahat[b] @ Y[b] + bias), Ahat built on the fly from gso+dinv.
// Block = one batch (64x64 @ 64x128); gso AND the Y tile are staged in LDS
// (Y's fragment gather is column-strided, so gather from LDS, not HBM).
// 8 waves cover 4 Mtiles x 2 N-halves.
// -------------------------------------------------------------------------
__global__ void __launch_bounds__(256)
gcn_aggregate(const float* __restrict__ gso, const float* __restrict__ dinv,
              const float* __restrict__ Y, const float* __restrict__ bias,
              float* __restrict__ H) {
  __shared__ float s_g[64 * 64];   // 16 KB
  __shared__ float s_y[64 * 128];  // 32 KB
  __shared__ float s_d[64];
  const int b   = blockIdx.x;
  const int tid = threadIdx.x;
  for (int i = tid; i < 64 * 64; i += 256) s_g[i] = gso[(size_t)b * 4096 + i];
  for (int i = tid; i < 64 * 128; i += 256) s_y[i] = Y[(size_t)b * 8192 + i];
  if (tid < 64) s_d[tid] = dinv[b * 64 + tid];
  __syncthreads();

  const int wave = tid >> 5;
  const int lane = tid & 31;
  const int mt = wave & 3;  // node-row tile
  const int nh = wave >> 2; // feature half (0: cols 0-63, 1: cols 64-127)

  v8f acc[4];
#pragma unroll
  for (int nt = 0; nt < 4; ++nt) acc[nt] = v8f_zero();

  const int m = mt * 16 + (lane & 15);
  for (int ks = 0; ks < 2; ++ks) {
    v16bf afrag; // Ahat(m,k) = d[m]*(gso[m,k]+I)*d[k]
#pragma unroll
    for (int e = 0; e < 16; ++e) {
      int k    = ks * 32 + a_kmap(lane, e);
      float a1 = s_g[m * 64 + k] + ((m == k) ? 1.0f : 0.0f);
      afrag[e] = (bf16_t)(s_d[m] * a1 * s_d[k]);
    }
#pragma unroll
    for (int nt = 0; nt < 4; ++nt) {
      const int n = (nh * 4 + nt) * 16 + (lane & 15);
      v16bf bfrag;
#pragma unroll
      for (int e = 0; e < 16; ++e) {
        int k    = ks * 32 + b_kmap(lane, e);
        bfrag[e] = (bf16_t)s_y[k * 128 + n];
      }
      acc[nt] = wmma_bf16(afrag, bfrag, acc[nt]);
    }
  }
#pragma unroll
  for (int nt = 0; nt < 4; ++nt) {
    const int n  = (nh * 4 + nt) * 16 + (lane & 15);
    const float bi = bias[n];
#pragma unroll
    for (int v = 0; v < 8; ++v) {
      int mm = mt * 16 + v + ((lane >= 16) ? 8 : 0);
      H[((size_t)b * 64 + mm) * 128 + n] = fmaxf(acc[nt][v] + bi, 0.0f);
    }
  }
}

// -------------------------------------------------------------------------
// out[m,o] = Hg[m,:] . Wo[o,:] + bo[o]   (N=5 head: trivial VALU work)
// -------------------------------------------------------------------------
__global__ void out_head(const float* __restrict__ Hg, const float* __restrict__ Wo,
                         const float* __restrict__ bo, float* __restrict__ outp, int M) {
  int i = blockIdx.x * blockDim.x + threadIdx.x;
  if (i >= M * 5) return;
  int m = i / 5, o = i % 5;
  const float4* hr = (const float4*)(Hg + (size_t)m * 128);
  const float4* wr = (const float4*)(Wo + (size_t)o * 128);
  float s = bo[o];
#pragma unroll
  for (int k = 0; k < 32; ++k) {
    float4 h = hr[k], w = wr[k];
    s += h.x * w.x + h.y * w.y + h.z * w.z + h.w * w.w;
  }
  outp[i] = s;
}

// -------------------------------------------------------------------------
extern "C" void kernel_launch(void* const* d_in, const int* in_sizes, int n_in,
                              void* d_out, int out_size, void* d_ws, size_t ws_size,
                              hipStream_t stream) {
  (void)in_sizes; (void)n_in; (void)out_size; (void)ws_size;
  const float* x      = (const float*)d_in[0];
  const float* gso    = (const float*)d_in[1];
  const float* w_c1   = (const float*)d_in[2];
  const float* b_c1   = (const float*)d_in[3];
  const float* g_bn1  = (const float*)d_in[4];
  const float* be_bn1 = (const float*)d_in[5];
  const float* w_c2   = (const float*)d_in[6];
  const float* b_c2   = (const float*)d_in[7];
  const float* g_bn2  = (const float*)d_in[8];
  const float* be_bn2 = (const float*)d_in[9];
  const float* w_c3   = (const float*)d_in[10];
  const float* b_c3   = (const float*)d_in[11];
  const float* g_bn3  = (const float*)d_in[12];
  const float* be_bn3 = (const float*)d_in[13];
  const float* w_fc   = (const float*)d_in[14];
  const float* b_fc   = (const float*)d_in[15];
  const float* w_g1   = (const float*)d_in[16];
  const float* b_g1   = (const float*)d_in[17];
  const float* w_g2   = (const float*)d_in[18];
  const float* b_g2   = (const float*)d_in[19];
  const float* w_out  = (const float*)d_in[20];
  const float* b_outp = (const float*)d_in[21];

  const int NIMG = 256 * 64;                              // B*N
  const size_t SZV = (size_t)NIMG * 128 * sizeof(float);  // 8 MiB unit

  char* ws    = (char*)d_ws;
  float* h1   = (float*)(ws + 0);        // [NIMG,32,4,4] = 4*SZV (dead after conv2)
  float* feat = (float*)(ws + 0);        // reuses h1 region
  float* y    = (float*)(ws + 1 * SZV);  // y1 then y2
  float* hg   = (float*)(ws + 2 * SZV);
  float* hg2  = (float*)(ws + 3 * SZV);
  float* h2   = (float*)(ws + 4 * SZV);  // [NIMG,64,2,2] = 2*SZV
  float* h3   = (float*)(ws + 6 * SZV);  // [NIMG,128]    = 1*SZV
  float* dinv = (float*)(ws + 7 * SZV);  // [NIMG]

  dinv_kernel<<<NIMG / 256, 256, 0, stream>>>(gso, dinv);

  conv_bn_relu_pool<3, 32, 8, 1>
      <<<NIMG / 4, 128, 0, stream>>>(x, w_c1, b_c1, g_bn1, be_bn1, h1, NIMG);
  conv_bn_relu_pool<32, 64, 4, 1>
      <<<NIMG / 4, 128, 0, stream>>>(h1, w_c2, b_c2, g_bn2, be_bn2, h2, NIMG);
  conv_bn_relu_pool<64, 128, 2, 4>
      <<<NIMG / 16, 128, 0, stream>>>(h2, w_c3, b_c3, g_bn3, be_bn3, h3, NIMG);

  gemm_rowmajor_wt<128, 128, true>
      <<<NIMG / 64, 128, 0, stream>>>(h3, w_fc, b_fc, feat, NIMG);

  gemm_rowmajor_wt<128, 128, false>
      <<<NIMG / 64, 128, 0, stream>>>(feat, w_g1, nullptr, y, NIMG);
  gcn_aggregate<<<256, 256, 0, stream>>>(gso, dinv, y, b_g1, hg);

  gemm_rowmajor_wt<128, 128, false>
      <<<NIMG / 64, 128, 0, stream>>>(hg, w_g2, nullptr, y, NIMG);
  gcn_aggregate<<<256, 256, 0, stream>>>(gso, dinv, y, b_g2, hg2);

  out_head<<<(NIMG * 5 + 255) / 256, 256, 0, stream>>>(hg2, w_out, b_outp,
                                                       (float*)d_out, NIMG);
}
